// Attention_87926570484489
// MI455X (gfx1250) — matrix operand order repaired and checked by reading, not compiled, not run on hardware
//
#include <hip/hip_runtime.h>

// ---------------- types ----------------
typedef __attribute__((ext_vector_type(16))) __bf16 v16bf;
typedef __attribute__((ext_vector_type(8)))  float  v8f;

struct U32x8 { uint4 a, b; };

__device__ inline v8f vzero() {
    v8f z = {0.f, 0.f, 0.f, 0.f, 0.f, 0.f, 0.f, 0.f};
    return z;
}

__device__ inline unsigned short bfbits(float f) {
    unsigned u = __builtin_bit_cast(unsigned, f);
    u += 0x7FFFu + ((u >> 16) & 1u);          // round-to-nearest-even
    return (unsigned short)(u >> 16);
}
__device__ inline __bf16 f2bf(float f) {
    unsigned short h = bfbits(f);
    return __builtin_bit_cast(__bf16, h);
}
__device__ inline __bf16 u16bf(unsigned short h) {
    return __builtin_bit_cast(__bf16, h);
}
__device__ inline unsigned pack2(float a, float b) {
    return (unsigned)bfbits(a) | ((unsigned)bfbits(b) << 16);
}
__device__ inline uint4 pack8(float4 f0, float4 f1) {
    uint4 u;
    u.x = pack2(f0.x, f0.y);
    u.y = pack2(f0.z, f0.w);
    u.z = pack2(f1.x, f1.y);
    u.w = pack2(f1.z, f1.w);
    return u;
}

// Build a 16-element bf16 fragment from two contiguous 16B chunks.
__device__ inline v16bf make_frag(const __bf16* p0, const __bf16* p1) {
    U32x8 t;
    t.a = *reinterpret_cast<const uint4*>(p0);
    t.b = *reinterpret_cast<const uint4*>(p1);
    return __builtin_bit_cast(v16bf, t);
}

__device__ inline v8f wmma_bf16(v16bf a, v16bf b, v8f c) {
    // D = A(16x32) * B(32x16) + C, fp32 accumulate
    return __builtin_amdgcn_wmma_f32_16x16x32_bf16(
        /*neg_a=*/false, a, /*neg_b=*/false, b,
        /*c_mod=*/(short)0, c, /*reuse_a=*/false, /*reuse_b=*/false);
}

// Async global->LDS copy of 32B (two b128 beats), tracked by ASYNCcnt.
// The instruction's INST_OFFSET is added to BOTH the LDS and global address.
__device__ inline void async_copy_32B(const __bf16* gsrc, void* lds_dst) {
    unsigned ldsaddr = (unsigned)(size_t)lds_dst;   // low 32 bits of flat = LDS offset
    asm volatile("global_load_async_to_lds_b128 %0, %1, off"
                 :: "v"(ldsaddr), "v"(gsrc) : "memory");
    asm volatile("global_load_async_to_lds_b128 %0, %1, off offset:16"
                 :: "v"(ldsaddr), "v"(gsrc) : "memory");
}
__device__ inline void wait_async0() {
    asm volatile("s_wait_asynccnt 0" ::: "memory");
}

// Wave index as a compiler-provable scalar (tid>>5 is wave-uniform, but the
// uniformity analysis can't see that; readfirstlane pins it to an SGPR so all
// derived tile offsets / pointer selects become SALU + scalar addressing).
__device__ inline int wave_id() {
    return __builtin_amdgcn_readfirstlane((int)(threadIdx.x >> 5));
}

// A-fragment (16x32, MxK): lane = row (lane&15), koff = lane<16 ? 0 : 16,
// elements 0..15 -> K=koff..koff+15 (contiguous, same as B).
// B-fragment (32x16, KxN): lane = col (lane&15), koff = lane<16 ? 0 : 16,
// elements 0..15 -> K=koff..koff+15 (contiguous).

#define SEQ   4096
#define CDIM  768
#define HEADS 12
#define HD    64
#define MROWS (2 * SEQ)          // 8192
#define SCALE 0.125f             // 64^-0.5

#define TK    32
#define LDSTR 40                 // LDS row stride (elements); 80B = multiple of 16B

// ================= QKV GEMM: Y = X @ Wqkv^T + b, scatter to Q/K/V[B,H,N,64] (bf16) ============
__global__ __launch_bounds__(256) void qkv_gemm_kernel(
    const float* __restrict__ X, const float* __restrict__ W, const float* __restrict__ bias,
    __bf16* __restrict__ Qw, __bf16* __restrict__ Kw, __bf16* __restrict__ Vw)
{
    __shared__ __bf16 Alds[128][LDSTR];   // X tile   [m][k]
    __shared__ __bf16 Blds[128][LDSTR];   // W rows   [n][k]  (== W^T columns, K-contiguous)

    const int tid  = threadIdx.x;
    const int lane = tid & 31;
    const int wid  = wave_id();           // SGPR
    const int m0 = blockIdx.y * 128;
    const int n0 = blockIdx.x * 128;
    const int wm = (wid & 3) * 32;        // scalar: 4 waves along M
    const int wn = (wid >> 2) * 64;       // scalar: 2 waves along N

    v8f acc[2][4];
    #pragma unroll
    for (int i = 0; i < 2; ++i)
        #pragma unroll
        for (int j = 0; j < 4; ++j) acc[i][j] = vzero();

    const int lr = tid >> 1;              // 0..127
    const int ks = (tid & 1) * 16;        // 0 or 16

    for (int k0 = 0; k0 < CDIM; k0 += TK) {
        // ---- cooperative fp32 -> bf16 tile staging ----
        const float4* sa4 = (const float4*)(X + (size_t)(m0 + lr) * CDIM + (k0 + ks));
        *(uint4*)&Alds[lr][ks]     = pack8(sa4[0], sa4[1]);
        *(uint4*)&Alds[lr][ks + 8] = pack8(sa4[2], sa4[3]);
        const float4* sb4 = (const float4*)(W + (size_t)(n0 + lr) * CDIM + (k0 + ks));
        *(uint4*)&Blds[lr][ks]     = pack8(sb4[0], sb4[1]);
        *(uint4*)&Blds[lr][ks + 8] = pack8(sb4[2], sb4[3]);
        __syncthreads();

        v16bf afr[2], bfr[4];
        {
            const int koffA = (lane < 16) ? 0 : 16;
            #pragma unroll
            for (int i = 0; i < 2; ++i) {
                const __bf16* row = &Alds[wm + i * 16 + (lane & 15)][0];
                afr[i] = make_frag(row + koffA, row + koffA + 8);
            }
            const int koffB = (lane < 16) ? 0 : 16;
            #pragma unroll
            for (int j = 0; j < 4; ++j) {
                const __bf16* row = &Blds[wn + j * 16 + (lane & 15)][0];
                bfr[j] = make_frag(row + koffB, row + koffB + 8);
            }
        }
        #pragma unroll
        for (int i = 0; i < 2; ++i)
            #pragma unroll
            for (int j = 0; j < 4; ++j)
                acc[i][j] = wmma_bf16(afr[i], bfr[j], acc[i][j]);
        __syncthreads();
    }

    // ---- epilogue: wave-uniform (now provably scalar) part/head/batch ----
    // This wave's 64 columns are 64-aligned; 768 % 64 == 0 and head width == 64,
    // so 'part' (q/k/v) and the head index are uniform across the wave.
    const int c0   = n0 + wn;             // scalar, 64-aligned
    const int part = c0 / CDIM;
    const int hh   = (c0 % CDIM) >> 6;    // scalar head index
    __bf16* dst    = (part == 0) ? Qw : (part == 1) ? Kw : Vw;
    const float sc = (part == 0) ? SCALE : 1.0f;
    const int bb   = m0 >> 12;            // scalar batch (128-row tile within one batch)
    const int half = lane >> 4;

    __bf16* base = dst + (size_t)(bb * HEADS + hh) * SEQ * HD;   // scalar base
    #pragma unroll
    for (int i = 0; i < 2; ++i) {
        #pragma unroll
        for (int r = 0; r < 8; ++r) {
            int m  = m0 + wm + i * 16 + r + 8 * half;
            int nn = m & (SEQ - 1);
            __bf16* rowp = base + (size_t)nn * HD;
            #pragma unroll
            for (int j = 0; j < 4; ++j) {
                int dd = j * 16 + (lane & 15);        // offset within head
                float v = acc[i][j][r] + bias[c0 + dd];
                rowp[dd] = f2bf(v * sc);
            }
        }
    }
}

// ================= Flash attention: 4 waves, 16 queries/wave, 32-key tiles ====================
#define KST 72   // K tile LDS stride (144B, multiple of 16B)
#define VST 40   // V^T tile LDS stride (80B)

__global__ __launch_bounds__(128) void attn_kernel(
    const __bf16* __restrict__ Qw, const __bf16* __restrict__ Kw,
    const __bf16* __restrict__ Vw, __bf16* __restrict__ AO)
{
    __shared__ __bf16 Klds[32][KST];       // [key][d]   (row-major like global)
    __shared__ __bf16 VldsT[64][VST];      // [d][key]   (transposed for PV B-frag)
    __shared__ __bf16 Plds[4][16][32];     // per-wave P tile (C-layout -> A-layout shuffle)

    const int tid  = threadIdx.x;
    const int lane = tid & 31;
    const int wid  = wave_id();            // SGPR
    const int bh   = blockIdx.y;           // 0..23
    const int b    = bh / HEADS, h = bh % HEADS;
    const int qbase = blockIdx.x * 64;

    const __bf16* Qp = Qw + (size_t)bh * SEQ * HD;
    const __bf16* Kp = Kw + (size_t)bh * SEQ * HD;
    const __bf16* Vp = Vw + (size_t)bh * SEQ * HD;

    // Q fragments for this wave's 16 queries (Q already scaled), D split 0..31 / 32..63
    const int koffA = (lane < 16) ? 0 : 16;
    const __bf16* qrow = Qp + (size_t)(qbase + wid * 16 + (lane & 15)) * HD;
    v16bf qa0 = make_frag(qrow + koffA,      qrow + koffA + 8);
    v16bf qa1 = make_frag(qrow + 32 + koffA, qrow + 32 + koffA + 8);

    float mr[8], lrow[8];
    v8f acc[4];
    #pragma unroll
    for (int r = 0; r < 8; ++r) { mr[r] = -1e30f; lrow[r] = 0.f; }
    #pragma unroll
    for (int j = 0; j < 4; ++j) acc[j] = vzero();

    const int crow = tid >> 2;             // 0..31 (key row for staging)
    const int cseg = (tid & 3) * 16;       // 0/16/32/48 (d segment)

    for (int kb = 0; kb < SEQ; kb += 32) {
        // ---- stage K via async global->LDS DMA (no VGPR round trip) ----
        async_copy_32B(Kp + (size_t)(kb + crow) * HD + cseg, &Klds[crow][cseg]);

        // ---- stage V transposed: 2x b128 global loads, register extract, b16 LDS stores ----
        {
            const __bf16* sv = Vp + (size_t)(kb + crow) * HD + cseg;
            uint4 w0 = *(const uint4*)sv;
            uint4 w1 = *(const uint4*)(sv + 8);
            unsigned w[8] = {w0.x, w0.y, w0.z, w0.w, w1.x, w1.y, w1.z, w1.w};
            #pragma unroll
            for (int i = 0; i < 8; ++i) {
                VldsT[cseg + 2 * i][crow]     = u16bf((unsigned short)(w[i] & 0xffffu));
                VldsT[cseg + 2 * i + 1][crow] = u16bf((unsigned short)(w[i] >> 16));
            }
        }
        if (kb + 32 < SEQ) {
            __builtin_prefetch(Kp + (size_t)(kb + 32 + crow) * HD, 0, 0);
            __builtin_prefetch(Vp + (size_t)(kb + 32 + crow) * HD, 0, 0);
        }
        wait_async0();            // this wave's async K writes are in LDS
        __syncthreads();          // all waves' K/V tiles visible

        // ---- S = Q * K^T : two 16-key halves, D accumulated over 2 WMMAs each ----
        v8f s[2];
        {
            const int koffB = (lane < 16) ? 0 : 16;
            #pragma unroll
            for (int kh = 0; kh < 2; ++kh) {
                const __bf16* krow = &Klds[kh * 16 + (lane & 15)][0];
                v16bf kb0 = make_frag(krow + koffB,      krow + koffB + 8);
                v16bf kb1 = make_frag(krow + 32 + koffB, krow + 32 + koffB + 8);
                v8f t = vzero();
                t = wmma_bf16(qa0, kb0, t);
                t = wmma_bf16(qa1, kb1, t);
                s[kh] = t;
            }
        }

        // ---- online softmax (row stats reduced across each 16-lane half-group) ----
        const int half = lane >> 4;
        #pragma unroll
        for (int r = 0; r < 8; ++r) {
            float smax = fmaxf(s[0][r], s[1][r]);
            #pragma unroll
            for (int off = 8; off >= 1; off >>= 1)
                smax = fmaxf(smax, __shfl_xor(smax, off, 32));
            float mnew = fmaxf(mr[r], smax);
            float corr = __expf(mr[r] - mnew);
            mr[r] = mnew;
            float p0 = __expf(s[0][r] - mnew);
            float p1 = __expf(s[1][r] - mnew);
            float rs = p0 + p1;
            #pragma unroll
            for (int off = 8; off >= 1; off >>= 1)
                rs += __shfl_xor(rs, off, 32);
            lrow[r] = lrow[r] * corr + rs;
            #pragma unroll
            for (int j = 0; j < 4; ++j) acc[j][r] *= corr;
            int prow = r + 8 * half;                 // C-layout row for this VGPR
            Plds[wid][prow][lane & 15]        = f2bf(p0);
            Plds[wid][prow][16 + (lane & 15)] = f2bf(p1);
        }

        // ---- O += P * V (P re-read in A-layout; wave-local LDS, DS pipe is in-order) ----
        {
            const __bf16* prow = &Plds[wid][lane & 15][0];
            v16bf pa = make_frag(prow + koffA, prow + koffA + 8);
            const int koffB = (lane < 16) ? 0 : 16;
            #pragma unroll
            for (int j = 0; j < 4; ++j) {
                const __bf16* vrow = &VldsT[j * 16 + (lane & 15)][0];
                v16bf vb = make_frag(vrow + koffB, vrow + koffB + 8);
                acc[j] = wmma_bf16(pa, vb, acc[j]);
            }
        }
        __syncthreads();
    }

    // ---- normalize (reciprocal once per row) and write [B, N, H*64] bf16 ----
    float linv[8];
    #pragma unroll
    for (int r = 0; r < 8; ++r) linv[r] = 1.0f / lrow[r];

    const int half = lane >> 4;
    #pragma unroll
    for (int j = 0; j < 4; ++j) {
        #pragma unroll
        for (int r = 0; r < 8; ++r) {
            int q = qbase + wid * 16 + r + 8 * half;
            int c = h * HD + j * 16 + (lane & 15);
            AO[(size_t)(b * SEQ + q) * CDIM + c] = f2bf(acc[j][r] * linv[r]);
        }
    }
}

// ================= Projection GEMM: out = AO @ proj_w^T + proj_b (fp32 out) ==================
__global__ __launch_bounds__(256) void proj_gemm_kernel(
    const __bf16* __restrict__ AO, const float* __restrict__ W,
    const float* __restrict__ bias, float* __restrict__ out)
{
    __shared__ __bf16 Alds[128][LDSTR];
    __shared__ __bf16 Blds[128][LDSTR];

    const int tid  = threadIdx.x;
    const int lane = tid & 31;
    const int wid  = wave_id();           // SGPR
    const int m0 = blockIdx.y * 128;
    const int n0 = blockIdx.x * 128;
    const int wm = (wid & 3) * 32;
    const int wn = (wid >> 2) * 64;

    v8f acc[2][4];
    #pragma unroll
    for (int i = 0; i < 2; ++i)
        #pragma unroll
        for (int j = 0; j < 4; ++j) acc[i][j] = vzero();

    const int lr = tid >> 1;
    const int ks = (tid & 1) * 16;

    for (int k0 = 0; k0 < CDIM; k0 += TK) {
        // A tile (already bf16): async global->LDS DMA
        async_copy_32B(AO + (size_t)(m0 + lr) * CDIM + (k0 + ks), &Alds[lr][ks]);
        // B tile: fp32 -> bf16 through VGPRs
        const float4* sb4 = (const float4*)(W + (size_t)(n0 + lr) * CDIM + (k0 + ks));
        *(uint4*)&Blds[lr][ks]     = pack8(sb4[0], sb4[1]);
        *(uint4*)&Blds[lr][ks + 8] = pack8(sb4[2], sb4[3]);
        wait_async0();
        __syncthreads();

        v16bf afr[2], bfr[4];
        {
            const int koffA = (lane < 16) ? 0 : 16;
            #pragma unroll
            for (int i = 0; i < 2; ++i) {
                const __bf16* row = &Alds[wm + i * 16 + (lane & 15)][0];
                afr[i] = make_frag(row + koffA, row + koffA + 8);
            }
            const int koffB = (lane < 16) ? 0 : 16;
            #pragma unroll
            for (int j = 0; j < 4; ++j) {
                const __bf16* row = &Blds[wn + j * 16 + (lane & 15)][0];
                bfr[j] = make_frag(row + koffB, row + koffB + 8);
            }
        }
        #pragma unroll
        for (int i = 0; i < 2; ++i)
            #pragma unroll
            for (int j = 0; j < 4; ++j)
                acc[i][j] = wmma_bf16(afr[i], bfr[j], acc[i][j]);
        __syncthreads();
    }

    const int half = lane >> 4;
    #pragma unroll
    for (int i = 0; i < 2; ++i) {
        #pragma unroll
        for (int j = 0; j < 4; ++j) {
            #pragma unroll
            for (int r = 0; r < 8; ++r) {
                int m = m0 + wm + i * 16 + r + 8 * half;
                int c = n0 + wn + j * 16 + (lane & 15);
                out[(size_t)m * CDIM + c] = acc[i][j][r] + bias[c];
            }
        }
    }
}

// ======================================= launcher ===========================================
extern "C" void kernel_launch(void* const* d_in, const int* in_sizes, int n_in,
                              void* d_out, int out_size, void* d_ws, size_t ws_size,
                              hipStream_t stream) {
    const float* x      = (const float*)d_in[0];
    const float* qkv_w  = (const float*)d_in[1];
    const float* qkv_b  = (const float*)d_in[2];
    const float* proj_w = (const float*)d_in[3];
    const float* proj_b = (const float*)d_in[4];
    float* out = (float*)d_out;

    const size_t head_elems = (size_t)2 * HEADS * SEQ * HD;   // 6,291,456 per tensor
    __bf16* Qw = (__bf16*)d_ws;
    __bf16* Kw = Qw + head_elems;
    __bf16* Vw = Kw + head_elems;
    __bf16* AO = Vw + head_elems;                              // [8192, 768] bf16

    // 1) QKV projection (M=8192, N=2304, K=768)
    qkv_gemm_kernel<<<dim3(2304 / 128, MROWS / 128), 256, 0, stream>>>(
        x, qkv_w, qkv_b, Qw, Kw, Vw);
    // 2) Flash attention: 64-query tiles x 24 (b,h) pairs
    attn_kernel<<<dim3(SEQ / 64, 2 * HEADS), 128, 0, stream>>>(Qw, Kw, Vw, AO);
    // 3) Output projection (M=8192, N=768, K=768)
    proj_gemm_kernel<<<dim3(CDIM / 128, MROWS / 128), 256, 0, stream>>>(
        AO, proj_w, proj_b, out);
}
